// NEAT_45878840655915
// MI455X (gfx1250) — matrix operand order repaired and checked by reading, not compiled, not run on hardware
//
#include <hip/hip_runtime.h>
#include <hip/hip_bf16.h>
#include <math.h>

// NEAT forward on MI455X (gfx1250, wave32).
// One workgroup (512 thr = 16 waves) per genome.
// LDS: dense f16 weight matrix W[col*256+row] (128KB) built once via
// ds_pk_add_f16 atomics, then 3 layers of 256x256 matvec done with
// v_wmma_f32_16x16x32_f16 (vector broadcast into all 16 B columns).

#define NEAT_B   4096
#define NEAT_N   256
#define NEAT_E   4096
#define NEAT_NI  64
#define NEAT_NO  8
#define NEAT_NL  4

typedef __attribute__((ext_vector_type(16))) _Float16 v16h;
typedef __attribute__((ext_vector_type(8)))  float    v8f;

__global__ __launch_bounds__(512)
void neat_forward_kernel(const float* __restrict__ inputs,
                         const int*   __restrict__ node_layer,
                         const int*   __restrict__ conn_in,
                         const int*   __restrict__ conn_out,
                         const float* __restrict__ conn_w,
                         const int*   __restrict__ conn_en,
                         float*       __restrict__ out) {
    __shared__ unsigned Wu[NEAT_N * NEAT_N / 2];   // 65536 f16 = 128KB, col-major
    __shared__ float    vals[NEAT_N];
    __shared__ float    agg[NEAT_N];
    __shared__ int      incnt[NEAT_N];
    __shared__ int      nlayer[NEAT_N];

    const int g   = blockIdx.x;
    const int tid = threadIdx.x;
    _Float16* W16 = (_Float16*)Wu;
    const unsigned wbase = (unsigned)(uintptr_t)&Wu[0];  // flat addr low 32 bits = LDS byte offset

    // ---- Phase 0: zero W (uint4 stores -> ds_store_b128), init node state ----
    {
        uint4* w4 = (uint4*)Wu;
        #pragma unroll
        for (int i = tid; i < (NEAT_N * NEAT_N / 8); i += 512) {
            w4[i] = make_uint4(0u, 0u, 0u, 0u);
        }
    }
    if (tid < NEAT_N) {
        vals[tid]   = (tid < NEAT_NI) ? inputs[(size_t)g * NEAT_NI + tid] : 0.0f;
        incnt[tid]  = 0;
        nlayer[tid] = node_layer[(size_t)g * NEAT_N + tid];
    }
    __syncthreads();

    // ---- Phase 1: stream edges once (coalesced 16B/lane), scatter into LDS ----
    {
        const int4*   ci4 = (const int4*)  (conn_in  + (size_t)g * NEAT_E);
        const int4*   co4 = (const int4*)  (conn_out + (size_t)g * NEAT_E);
        const float4* cw4 = (const float4*)(conn_w   + (size_t)g * NEAT_E);
        const int4*   ce4 = (const int4*)  (conn_en  + (size_t)g * NEAT_E);

        #pragma unroll
        for (int i = tid; i < NEAT_E / 4; i += 512) {
            const int4   ci = ci4[i];
            const int4   co = co4[i];
            const float4 cw = cw4[i];
            const int4   ce = ce4[i];
            #define NEAT_PROC(CI, CO, WW, EN)                                        \
                if (EN) {                                                            \
                    const unsigned col = (unsigned)(CI) & 255u;                      \
                    const unsigned row = (unsigned)(CO) & 255u;                      \
                    const unsigned idx = col * 256u + row;                           \
                    union { _Float16 h[2]; unsigned u; } pk;                         \
                    pk.u = 0u;                                                       \
                    pk.h[idx & 1u] = (_Float16)(WW);                                 \
                    const unsigned addr = wbase + ((idx >> 1) << 2);                 \
                    asm volatile("ds_pk_add_f16 %0, %1" :: "v"(addr), "v"(pk.u)      \
                                 : "memory");                                        \
                    atomicAdd(&incnt[row], 1);                                       \
                }
            NEAT_PROC(ci.x, co.x, cw.x, ce.x)
            NEAT_PROC(ci.y, co.y, cw.y, ce.y)
            NEAT_PROC(ci.z, co.z, cw.z, ce.z)
            NEAT_PROC(ci.w, co.w, cw.w, ce.w)
            #undef NEAT_PROC
        }
    }
    // compiler can't see the asm DS atomics -> drain DScnt manually before barrier
    asm volatile("s_wait_dscnt 0" ::: "memory");
    __syncthreads();

    // ---- Phase 2: 3 layers of WMMA matvec + masked tanh update ----
    const int lane = tid & 31;
    const int wv   = tid >> 5;          // wave id = 16-row group, 0..15
    const int hlf  = (lane >> 4) & 1;   // half-wave select
    const int row  = (wv << 4) + (lane & 15);

    for (int layer = 1; layer < NEAT_NL; ++layer) {
        v8f c = {0.f, 0.f, 0.f, 0.f, 0.f, 0.f, 0.f, 0.f};
        #pragma unroll
        for (int kb = 0; kb < 8; ++kb) {
            const int K0 = kb << 5;
            v16h a, b;
            #pragma unroll
            for (int j = 0; j < 16; ++j) {
                // A (16x32, MxK): M = lane%16 (-> row), K = K0 + j + 8*hlf + 8*(j/8)
                const int kA = K0 + j + (hlf << 3) + ((j >> 3) << 3);
                a[j] = W16[(kA << 8) + row];              // col-major: conflict-free
                // B (32x16, KxN): all 16 columns = vals (broadcast), K = K0 + j + 8*hlf + 8*(j/8)
                const int kB = K0 + j + (hlf << 3) + ((j >> 3) << 3);
                b[j] = (_Float16)vals[kB];
            }
            c = __builtin_amdgcn_wmma_f32_16x16x32_f16(
                    false, a, false, b, (short)0, c, false, false);
        }
        // D: lane (N=lane%16), reg r -> M = r + 8*hlf. Column 0 lanes publish.
        if ((lane & 15) == 0) {
            #pragma unroll
            for (int r = 0; r < 8; ++r) {
                agg[(wv << 4) + (hlf << 3) + r] = c[r];
            }
        }
        __syncthreads();
        if (tid < NEAT_N) {
            if (nlayer[tid] == layer && incnt[tid] > 0) {
                vals[tid] = tanhf(agg[tid]);
            }
        }
        __syncthreads();
    }

    // ---- Phase 3: write outputs ----
    if (tid < NEAT_NO) {
        out[(size_t)g * NEAT_NO + tid] = vals[NEAT_NI + tid];
    }
}

extern "C" void kernel_launch(void* const* d_in, const int* in_sizes, int n_in,
                              void* d_out, int out_size, void* d_ws, size_t ws_size,
                              hipStream_t stream) {
    (void)in_sizes; (void)n_in; (void)out_size; (void)d_ws; (void)ws_size;
    const float* inputs     = (const float*)d_in[0];
    const int*   node_layer = (const int*)  d_in[1];
    const int*   conn_in    = (const int*)  d_in[2];
    const int*   conn_out   = (const int*)  d_in[3];
    const float* conn_w     = (const float*)d_in[4];
    const int*   conn_en    = (const int*)  d_in[5];
    float*       out        = (float*)d_out;

    neat_forward_kernel<<<dim3(NEAT_B), dim3(512), 0, stream>>>(
        inputs, node_layer, conn_in, conn_out, conn_w, conn_en, out);
}
